// GraphDiffusionEmbedding_20615843020928
// MI455X (gfx1250) — matrix operand order repaired and checked by reading, not compiled, not run on hardware
//
#include <hip/hip_runtime.h>
#include <hip/hip_bf16.h>

// ---------------------------------------------------------------------------
// GraphDiffusionEmbedding for MI455X (gfx1250): bf16 WMMA GEMM pipeline.
// ---------------------------------------------------------------------------

typedef __attribute__((ext_vector_type(16))) __bf16 v16bf;
typedef __attribute__((ext_vector_type(8)))  __bf16 v8bf;
typedef __attribute__((ext_vector_type(8)))  float  v8f;

#define N_NODES   200000
#define N_EDGETBL 500000
#define EDGES     20000
#define KSEL      20
#define DIM       128
#define DTIME     100
#define CIN       356      // D + DE + DT
#define KP1       384      // CIN padded to mult of 32 (12 k-tiles)
#define F1        512      // 32 n-tiles
#define F2        128      // 8 n-tiles
#define EPB       4        // edges per block (tppr): 4*20 = 80 rows = 5 M-tiles
#define ROWS_T    80
#define SX        392      // X row stride (bf16), 384+8 pad -> bank-safe
#define S1        520      // H1 row stride (bf16), 512+8 pad
#define S2        136      // H2 row stride (f32), 128+8 pad
#define SXS       136      // source X stride (bf16), 128+8 pad
#define SMEM_TPPR (80*520*2 + 80*392*2 + (80+80+80+4)*4)   // 146896 B
#define SMEM_SRC  (64*520*2 + 64*136*2)                    // 83968 B

__device__ __forceinline__ unsigned short f2bf(float f) {
  unsigned int u = __builtin_bit_cast(unsigned int, f);
  u += 0x7FFFu + ((u >> 16) & 1u);            // round-to-nearest-even
  return (unsigned short)(u >> 16);
}
__device__ __forceinline__ float bf2f(unsigned short h) {
  return __builtin_bit_cast(float, (unsigned int)h << 16);
}
__device__ __forceinline__ void store_bf4(unsigned short* p, float a, float b,
                                          float c, float d) {
  uint2 v;
  v.x = (unsigned)f2bf(a) | ((unsigned)f2bf(b) << 16);
  v.y = (unsigned)f2bf(c) | ((unsigned)f2bf(d) << 16);
  *reinterpret_cast<uint2*>(p) = v;
}
__device__ __forceinline__ v8f zerov8() {
  v8f z;
  #pragma unroll
  for (int i = 0; i < 8; ++i) z[i] = 0.f;
  return z;
}
// A fragment: 16-bit A 16x32 layout. rowp -> &X[row*stride + kbase + lh*8].
__device__ __forceinline__ v16bf load_afrag(const unsigned short* rowp) {
  union { v16bf v; v8bf h[2]; } u;
  u.h[0] = *reinterpret_cast<const v8bf*>(rowp);        // K = kbase+lh*8 .. +7
  u.h[1] = *reinterpret_cast<const v8bf*>(rowp + 16);   // K = kbase+16+lh*8 ..
  return u.v;
}
__device__ __forceinline__ v8f wmma_bf16(v16bf a, v16bf b, v8f c) {
  return __builtin_amdgcn_wmma_f32_16x16x32_bf16(false, a, false, b,
                                                 (short)0, c, false, false);
}

// ---------------------------------------------------------------------------
// Weight pre-swizzle: W[K x N] fp32 row-major -> bf16 B-fragment tiles.
// Tile (kt, nt) stored at element ((nt*KT + kt)*512 + lane*16):
//   lane l<16 : col nt*16+l,  K = kt*32 + 0..7, 16..23
//   lane l>=16: col nt*16+l-16, K = kt*32 + 8..15, 24..31
// ---------------------------------------------------------------------------
__global__ void swz_kernel(const float* __restrict__ src,
                           unsigned short* __restrict__ dst,
                           int Ksrc, int Kp, int N) {
  const int t = blockIdx.x * 256 + threadIdx.x;
  if (t >= Kp * N) return;
  const int j = t & 15, l = (t >> 4) & 31, tile = t >> 9;
  const int KT = Kp >> 5;
  const int kt = tile % KT, nt = tile / KT;
  const int n = nt * 16 + (l & 15);
  const int k = kt * 32 + (l >> 4) * 8 + (j & 7) + (j >> 3) * 16;
  dst[t] = f2bf((k < Ksrc) ? src[k * N + n] : 0.f);
}

// Fold BatchNorm(eval) + fc1s bias into per-column affine.
__global__ void bnfold_kernel(const float* __restrict__ g, const float* __restrict__ b,
                              const float* __restrict__ m, const float* __restrict__ v,
                              const float* __restrict__ fc1s_b,
                              float* __restrict__ sc, float* __restrict__ sh) {
  const int c = blockIdx.x * 256 + threadIdx.x;
  if (c >= F1) return;
  const float s = g[c] * rsqrtf(v[c] + 1e-5f);
  sc[c] = s;
  sh[c] = (fc1s_b[c] - m[c]) * s + b[c];
}

// ---------------------------------------------------------------------------
// TPPR branch kernel: block = 4 edges (80 rows), grid = (5000, 2).
// ---------------------------------------------------------------------------
__global__ __launch_bounds__(256) void tppr_kernel(
    const float* __restrict__ memory, const float* __restrict__ edge_tbl,
    const float* __restrict__ sel_dt, const float* __restrict__ sel_w,
    const float* __restrict__ time_w, const float* __restrict__ time_b,
    const unsigned short* __restrict__ W1, const float* __restrict__ fc1_b,
    const float* __restrict__ ln_g, const float* __restrict__ ln_b,
    const unsigned short* __restrict__ W2, const float* __restrict__ fc2_b,
    const int* __restrict__ sel_nodes, const int* __restrict__ sel_edges,
    float* __restrict__ out) {
  extern __shared__ char smem[];
  unsigned short* H1 = (unsigned short*)smem;                 // [80][520] bf16
  char* pool = smem + ROWS_T * S1 * 2;                        // X and H2 alias
  unsigned short* Xs = (unsigned short*)pool;                 // [80][392] bf16
  float* H2 = (float*)pool;                                   // [80][136] f32
  float* rsum = (float*)(pool + ROWS_T * SX * 2);             // [80]
  float* rsq  = rsum + ROWS_T;                                // [80]
  float* wsel = rsq + ROWS_T;                                 // [80]
  float* ssum = wsel + ROWS_T;                                // [4]

  const int tid = threadIdx.x;
  const int w = tid >> 5, lane = tid & 31;
  const int lh = lane >> 4, ll = lane & 15;
  const int branch = blockIdx.y;
  const int e0 = blockIdx.x * EPB;

  if (tid < ROWS_T) { rsum[tid] = 0.f; rsq[tid] = 0.f; }

  // ---- Phase A: gather + time-encode into bf16 X tile --------------------
  for (int r = w; r < ROWS_T; r += 8) {
    const int le = r / KSEL, kk = r - le * KSEL;
    const int sidx = (branch * EDGES + (e0 + le)) * KSEL + kk;
    const int node = sel_nodes[sidx];
    const int edge = sel_edges[sidx];
    const float dt = sel_dt[sidx];
    unsigned short* xr = Xs + r * SX;
    float4 nf = reinterpret_cast<const float4*>(memory + (size_t)node * DIM)[lane];
    store_bf4(xr + lane * 4, nf.x, nf.y, nf.z, nf.w);
    float4 ef = reinterpret_cast<const float4*>(edge_tbl + (size_t)edge * DIM)[lane];
    store_bf4(xr + DIM + lane * 4, ef.x, ef.y, ef.z, ef.w);
    if (lane < 25) {                                  // DT=100 cols
      const int c = lane * 4;
      float t0 = cosf(dt * time_w[c]     + time_b[c]);
      float t1 = cosf(dt * time_w[c + 1] + time_b[c + 1]);
      float t2 = cosf(dt * time_w[c + 2] + time_b[c + 2]);
      float t3 = cosf(dt * time_w[c + 3] + time_b[c + 3]);
      store_bf4(xr + 256 + c, t0, t1, t2, t3);
    } else {                                          // pad cols 356..383
      store_bf4(xr + 256 + lane * 4, 0.f, 0.f, 0.f, 0.f);
    }
    if (lane == 0) wsel[r] = sel_w[sidx];
  }
  __syncthreads();

  // ---- GEMM1: X[80x384] @ W1[384x512], raw bf16 + LN stats ---------------
  const v16bf* W1v = reinterpret_cast<const v16bf*>(W1);
  for (int m = 0; m < 5; ++m) {
    const int mbase = m * 16;
    float s[8], q[8];
    #pragma unroll
    for (int v = 0; v < 8; ++v) { s[v] = 0.f; q[v] = 0.f; }
    for (int ns = 0; ns < 4; ++ns) {
      const int nt = w + ns * 8;
      v8f acc = zerov8();
      const unsigned short* arow = Xs + (mbase + ll) * SX + lh * 8;
      #pragma unroll
      for (int kt = 0; kt < 12; ++kt) {
        v16bf a = load_afrag(arow + kt * 32);
        v16bf b = W1v[(nt * 12 + kt) * 32 + lane];
        acc = wmma_bf16(a, b, acc);
      }
      const int col = nt * 16 + ll;
      const float bias = fc1_b[col];
      #pragma unroll
      for (int v = 0; v < 8; ++v) {
        const float f = acc[v] + bias;
        s[v] += f; q[v] += f * f;
        H1[(mbase + v + lh * 8) * S1 + col] = f2bf(f);
      }
    }
    #pragma unroll
    for (int st = 1; st < 16; st <<= 1) {
      #pragma unroll
      for (int v = 0; v < 8; ++v) {
        s[v] += __shfl_xor(s[v], st, 32);
        q[v] += __shfl_xor(q[v], st, 32);
      }
    }
    if (ll == 0) {
      #pragma unroll
      for (int v = 0; v < 8; ++v) {
        const int row = mbase + v + lh * 8;
        atomicAdd(&rsum[row], s[v]);
        atomicAdd(&rsq[row], q[v]);
      }
    }
  }
  __syncthreads();

  // ---- LayerNorm + ReLU in place on H1 -----------------------------------
  if (tid < ROWS_T) {
    const float mu = rsum[tid] * (1.f / 512.f);
    const float var = rsq[tid] * (1.f / 512.f) - mu * mu;
    rsum[tid] = mu;
    rsq[tid] = rsqrtf(var + 1e-5f);
  }
  if (tid >= 96 && tid < 96 + EPB) {
    const int le = tid - 96;
    float sv = 0.f;
    for (int kk = 0; kk < KSEL; ++kk) sv += wsel[le * KSEL + kk];
    ssum[le] = sv;
  }
  __syncthreads();
  for (int t = tid; t < ROWS_T * 128; t += 256) {
    const int r = t >> 7;
    const int c = (t & 127) << 2;
    const float mu = rsum[r], rs = rsq[r];
    unsigned short* hp = H1 + r * S1 + c;
    uint2 pv = *reinterpret_cast<uint2*>(hp);
    float x0 = bf2f((unsigned short)(pv.x & 0xFFFF));
    float x1 = bf2f((unsigned short)(pv.x >> 16));
    float x2 = bf2f((unsigned short)(pv.y & 0xFFFF));
    float x3 = bf2f((unsigned short)(pv.y >> 16));
    const float4 g4 = reinterpret_cast<const float4*>(ln_g)[c >> 2];
    const float4 b4 = reinterpret_cast<const float4*>(ln_b)[c >> 2];
    x0 = fmaxf((x0 - mu) * rs * g4.x + b4.x, 0.f);
    x1 = fmaxf((x1 - mu) * rs * g4.y + b4.y, 0.f);
    x2 = fmaxf((x2 - mu) * rs * g4.z + b4.z, 0.f);
    x3 = fmaxf((x3 - mu) * rs * g4.w + b4.w, 0.f);
    store_bf4(hp, x0, x1, x2, x3);
  }
  __syncthreads();

  // ---- GEMM2: H1[80x512] @ W2[512x128] -> H2 (aliases dead X) ------------
  const v16bf* W2v = reinterpret_cast<const v16bf*>(W2);
  {
    const int nt = w;
    const int col = nt * 16 + ll;
    const float bias = fc2_b[col];
    for (int m = 0; m < 5; ++m) {
      v8f acc = zerov8();
      const unsigned short* arow = H1 + (m * 16 + ll) * S1 + lh * 8;
      #pragma unroll
      for (int kt = 0; kt < 16; ++kt) {
        v16bf a = load_afrag(arow + kt * 32);
        v16bf b = W2v[(nt * 16 + kt) * 32 + lane];
        acc = wmma_bf16(a, b, acc);
      }
      #pragma unroll
      for (int v = 0; v < 8; ++v)
        H2[(m * 16 + v + lh * 8) * S2 + col] = acc[v] + bias;
    }
  }
  __syncthreads();

  // ---- Weighted reduction over K=20, write output slice ------------------
  for (int t = tid; t < EPB * F2; t += 256) {
    const int le = t >> 7, f = t & 127;
    const float sv = ssum[le];
    float acc = 0.f;
    if (sv > 0.f) {
      #pragma unroll
      for (int kk = 0; kk < KSEL; ++kk)
        acc += H2[(le * KSEL + kk) * S2 + f] * wsel[le * KSEL + kk];
      acc /= sv;
    }
    out[(size_t)(e0 + le) * (3 * F2) + (branch + 1) * F2 + f] = acc;
  }
}

// ---------------------------------------------------------------------------
// Source branch kernel: block = 64 rows, grid = 313.
// ---------------------------------------------------------------------------
__global__ __launch_bounds__(256) void src_kernel(
    const float* __restrict__ memory, const int* __restrict__ src_nodes,
    const unsigned short* __restrict__ W1S, const float* __restrict__ bnsc,
    const float* __restrict__ bnsh, const unsigned short* __restrict__ W2S,
    const float* __restrict__ fc2s_b, float* __restrict__ out) {
  extern __shared__ char smem[];
  unsigned short* H1 = (unsigned short*)smem;                  // [64][520]
  unsigned short* Xs = (unsigned short*)(smem + 64 * S1 * 2);  // [64][136]
  const int tid = threadIdx.x;
  const int w = tid >> 5, lane = tid & 31;
  const int lh = lane >> 4, ll = lane & 15;
  const int r0 = blockIdx.x * 64;

  for (int r = w; r < 64; r += 8) {
    unsigned short* xr = Xs + r * SXS;
    const int rg = r0 + r;
    if (rg < EDGES) {
      const int node = src_nodes[rg];
      float4 nf = reinterpret_cast<const float4*>(memory + (size_t)node * DIM)[lane];
      store_bf4(xr + lane * 4, nf.x, nf.y, nf.z, nf.w);
    } else {
      store_bf4(xr + lane * 4, 0.f, 0.f, 0.f, 0.f);
    }
    if (lane < 2) store_bf4(xr + DIM + lane * 4, 0.f, 0.f, 0.f, 0.f);
  }
  __syncthreads();

  // GEMM1: X[64x128] @ W1S[128x512], fused BN affine + ReLU.
  const v16bf* W1v = reinterpret_cast<const v16bf*>(W1S);
  for (int m = 0; m < 4; ++m) {
    for (int ns = 0; ns < 4; ++ns) {
      const int nt = w + ns * 8;
      v8f acc = zerov8();
      const unsigned short* arow = Xs + (m * 16 + ll) * SXS + lh * 8;
      #pragma unroll
      for (int kt = 0; kt < 4; ++kt) {
        v16bf a = load_afrag(arow + kt * 32);
        v16bf b = W1v[(nt * 4 + kt) * 32 + lane];
        acc = wmma_bf16(a, b, acc);
      }
      const int col = nt * 16 + ll;
      const float sc = bnsc[col], sh = bnsh[col];
      #pragma unroll
      for (int v = 0; v < 8; ++v) {
        const float f = fmaxf(acc[v] * sc + sh, 0.f);
        H1[(m * 16 + v + lh * 8) * S1 + col] = f2bf(f);
      }
    }
  }
  __syncthreads();

  // GEMM2: H1[64x512] @ W2S[512x128] -> out cols 0..127.
  const v16bf* W2v = reinterpret_cast<const v16bf*>(W2S);
  {
    const int nt = w;
    const int col = nt * 16 + ll;
    const float bias = fc2s_b[col];
    for (int m = 0; m < 4; ++m) {
      v8f acc = zerov8();
      const unsigned short* arow = H1 + (m * 16 + ll) * S1 + lh * 8;
      #pragma unroll
      for (int kt = 0; kt < 16; ++kt) {
        v16bf a = load_afrag(arow + kt * 32);
        v16bf b = W2v[(nt * 16 + kt) * 32 + lane];
        acc = wmma_bf16(a, b, acc);
      }
      #pragma unroll
      for (int v = 0; v < 8; ++v) {
        const int rg = r0 + m * 16 + v + lh * 8;
        if (rg < EDGES) out[(size_t)rg * (3 * F2) + col] = acc[v] + bias;
      }
    }
  }
}

// ---------------------------------------------------------------------------
extern "C" void kernel_launch(void* const* d_in, const int* in_sizes, int n_in,
                              void* d_out, int out_size, void* d_ws, size_t ws_size,
                              hipStream_t stream) {
  (void)in_sizes; (void)n_in; (void)out_size; (void)ws_size;
  const float* memory   = (const float*)d_in[0];
  const float* edge_tbl = (const float*)d_in[1];
  const float* sel_dt   = (const float*)d_in[2];
  const float* sel_w    = (const float*)d_in[3];
  const float* time_w   = (const float*)d_in[4];
  const float* time_b   = (const float*)d_in[5];
  const float* fc1_w    = (const float*)d_in[6];
  const float* fc1_b    = (const float*)d_in[7];
  const float* ln1_g    = (const float*)d_in[8];
  const float* ln1_b    = (const float*)d_in[9];
  const float* fc2_w    = (const float*)d_in[10];
  const float* fc2_b    = (const float*)d_in[11];
  const float* fc1s_w   = (const float*)d_in[12];
  const float* fc1s_b   = (const float*)d_in[13];
  const float* bn_g     = (const float*)d_in[14];
  const float* bn_b     = (const float*)d_in[15];
  const float* bn_m     = (const float*)d_in[16];
  const float* bn_v     = (const float*)d_in[17];
  const float* fc2s_w   = (const float*)d_in[18];
  const float* fc2s_b   = (const float*)d_in[19];
  const int* source_nodes = (const int*)d_in[20];
  const int* sel_nodes    = (const int*)d_in[21];
  const int* sel_edges    = (const int*)d_in[22];
  float* out = (float*)d_out;

  // Workspace layout (~792 KB): pre-swizzled bf16 weights + folded BN.
  unsigned short* W1  = (unsigned short*)d_ws;          // 384*512
  unsigned short* W2  = W1 + 384 * 512;                 // 512*128
  unsigned short* W1S = W2 + 512 * 128;                 // 128*512
  unsigned short* W2S = W1S + 128 * 512;                // 512*128
  float* BNSC = (float*)(W2S + 512 * 128);              // 512
  float* BNSH = BNSC + 512;                             // 512

  swz_kernel<<<(384 * 512 + 255) / 256, 256, 0, stream>>>(fc1_w, W1, 356, 384, 512);
  swz_kernel<<<(512 * 128 + 255) / 256, 256, 0, stream>>>(fc2_w, W2, 512, 512, 128);
  swz_kernel<<<(128 * 512 + 255) / 256, 256, 0, stream>>>(fc1s_w, W1S, 128, 128, 512);
  swz_kernel<<<(512 * 128 + 255) / 256, 256, 0, stream>>>(fc2s_w, W2S, 512, 512, 128);
  bnfold_kernel<<<2, 256, 0, stream>>>(bn_g, bn_b, bn_m, bn_v, fc1s_b, BNSC, BNSH);

  hipFuncSetAttribute((const void*)src_kernel,
                      hipFuncAttributeMaxDynamicSharedMemorySize, SMEM_SRC);
  hipFuncSetAttribute((const void*)tppr_kernel,
                      hipFuncAttributeMaxDynamicSharedMemorySize, SMEM_TPPR);

  src_kernel<<<(EDGES + 63) / 64, 256, SMEM_SRC, stream>>>(
      memory, source_nodes, W1S, BNSC, BNSH, W2S, fc2s_b, out);
  tppr_kernel<<<dim3(EDGES / EPB, 2), 256, SMEM_TPPR, stream>>>(
      memory, edge_tbl, sel_dt, sel_w, time_w, time_b,
      W1, fc1_b, ln1_g, ln1_b, W2, fc2_b, sel_nodes, sel_edges, out);
}
